// KVCompressorVarlen_47845935677693
// MI455X (gfx1250) — matrix operand order repaired and checked by reading, not compiled, not run on hardware
//
#include <hip/hip_runtime.h>
#include <hip/hip_bf16.h>
#include <stdint.h>

#define NUM_HEADS    4
#define HEAD_DIM     128
#define BLOCK_SIZE   32
#define BLOCK_STRIDE 16
#define KTOT         (BLOCK_SIZE * HEAD_DIM)   // 4096
#define KC_STEPS     (KTOT / 32)               // 128 K-chunks of 32
#define SWZ_ELEMS    (KC_STEPS * 8 * 32 * 16)  // per-matrix swizzled bf16 count

#define AS1 __attribute__((address_space(1)))

typedef __attribute__((ext_vector_type(16))) __bf16   v16bf;
typedef __attribute__((ext_vector_type(8)))  float    v8f;
typedef __attribute__((ext_vector_type(4)))  float    f32x4;
typedef __attribute__((ext_vector_type(4)))  unsigned u32x4;

struct BU { union { v16bf v; u32x4 q[2]; }; };

// ---------------------------------------------------------------------------
// Kernel 1: cu_out (third output -> tail of d_out) + gather base table.
// ---------------------------------------------------------------------------
__global__ void prep_kernel(const int* __restrict__ cu, int nseq, int total_out,
                            int* __restrict__ base, int* __restrict__ outTail) {
  __shared__ int sCuOut[64];                 // supports up to 63 sequences
  if (threadIdx.x == 0) {
    int acc = 0;
    sCuOut[0] = 0;
    for (int b = 0; b < nseq; ++b) {
      int n  = cu[b + 1] - cu[b];
      int ol = (n - BLOCK_SIZE) / BLOCK_STRIDE;
      if (ol < 0) ol = 0;
      acc += ol;
      sCuOut[b + 1] = acc;
    }
  }
  __syncthreads();
  for (int i = threadIdx.x; i <= nseq; i += blockDim.x)
    outTail[i] = sCuOut[i];
  for (int o = threadIdx.x; o < total_out; o += blockDim.x) {
    int b = 0;
    while (b + 1 < nseq && sCuOut[b + 1] <= o) ++b;
    base[o] = cu[b] + (o - sCuOut[b]) * BLOCK_STRIDE;
  }
}

// ---------------------------------------------------------------------------
// Kernel 2: w_k / w_v (f32) -> bf16 pre-swizzled per-lane B-fragment order:
// [mat][kc][ntile][lane][16 halfwords]; halfword idx holds B element
// (K = (lane>=16)*16 + idx, N = ntile*16 + (lane&15)).
// ---------------------------------------------------------------------------
__global__ void swizzle_w_kernel(const float* __restrict__ wk,
                                 const float* __restrict__ wv,
                                 __bf16* __restrict__ wsW) {
  int p = blockIdx.x * blockDim.x + threadIdx.x;
  if (p >= 2 * SWZ_ELEMS) return;
  int mat  = p / SWZ_ELEMS;
  int q    = p - mat * SWZ_ELEMS;
  int kc   = q >> 12;
  int rem  = q & 4095;
  int nt   = rem >> 9;
  int rem2 = rem & 511;
  int lane = rem2 >> 4;
  int idx  = rem2 & 15;
  int t    = ((lane >> 4) << 4) + idx;        // K within chunk
  int e    = nt * 16 + (lane & 15);           // N column
  const float* w = mat ? wv : wk;
  wsW[p] = (__bf16)w[(size_t)(kc * 32 + t) * HEAD_DIM + e];
}

// ---------------------------------------------------------------------------
// Build one A fragment from 16 f32s (global address space).
// p = lane's row pointer (dOff = hi*16 folded in) + q*32 floats.
// Elements 0..15 <- floats [0..15] (K = q*32 + hi*16 + 0..15)
// ---------------------------------------------------------------------------
__device__ __forceinline__ v16bf make_A(const AS1 float* p) {
  f32x4 f0 = *(const AS1 f32x4*)(p);
  f32x4 f1 = *(const AS1 f32x4*)(p + 4);
  f32x4 f2 = *(const AS1 f32x4*)(p + 8);
  f32x4 f3 = *(const AS1 f32x4*)(p + 12);
  v16bf a;
  a[0]  = (__bf16)f0.x; a[1]  = (__bf16)f0.y;
  a[2]  = (__bf16)f0.z; a[3]  = (__bf16)f0.w;
  a[4]  = (__bf16)f1.x; a[5]  = (__bf16)f1.y;
  a[6]  = (__bf16)f1.z; a[7]  = (__bf16)f1.w;
  a[8]  = (__bf16)f2.x; a[9]  = (__bf16)f2.y;
  a[10] = (__bf16)f2.z; a[11] = (__bf16)f2.w;
  a[12] = (__bf16)f3.x; a[13] = (__bf16)f3.y;
  a[14] = (__bf16)f3.z; a[15] = (__bf16)f3.w;
  return a;
}

// ---------------------------------------------------------------------------
// Kernel 3: main GEMM. One wave owns a 64(o) x 64(e) tile for one head,
// k-path or v-path. Row-outer loop; 4 K-chunks per row with immediate
// offsets; A fragments ping-ponged to kill WMMA->VALU WAR hazards.
// ---------------------------------------------------------------------------
template <bool SWZ>
__global__ __launch_bounds__(32)
void kv_compress_wmma(const float* __restrict__ kin,
                      const float* __restrict__ vin,
                      const float* __restrict__ wk,
                      const float* __restrict__ wv,
                      const int*   __restrict__ base,
                      const __bf16* __restrict__ wsW,
                      float* __restrict__ outc,
                      int total_out) {
  const int lane  = threadIdx.x;        // 0..31
  const int h     = blockIdx.y;
  const int zc    = blockIdx.z;         // 0..3
  const int which = zc >> 1;            // 0 = k-path, 1 = v-path
  const int ng    = zc & 1;             // N half: e in [ng*64, ng*64+64)

  const AS1 float* in = (const AS1 float*)(which ? vin : kin);
  const AS1 float* w  = (const AS1 float*)(which ? wv : wk);
  const AS1 __bf16* wsz =
      (const AS1 __bf16*)wsW + (size_t)which * SWZ_ELEMS;
  AS1 float* out = (AS1 float*)outc +
                   (size_t)which * ((size_t)total_out * NUM_HEADS * HEAD_DIM);
  const AS1 int* gbase = (const AS1 int*)base;

  const int oBase = blockIdx.x * 64;
  const int lrow  = lane & 15;
  const int hi    = lane >> 4;

  // Per-lane A row pointers for the 4 M-subtiles (clamped, EXEC stays full).
  const AS1 float* aPtr[4];
#pragma unroll
  for (int s = 0; s < 4; ++s) {
    int oA = oBase + s * 16 + lrow;
    if (oA > total_out - 1) oA = total_out - 1;
    aPtr[s] = in + (size_t)gbase[oA] * (NUM_HEADS * HEAD_DIM) +
              (size_t)h * HEAD_DIM + hi * 16;
  }

  // Per-lane B base (swizzled path): fragment (kc, ng*4+ntl) lives at
  // byte offset kc*8192 + (ng*4+ntl)*1024 + lane*32.
  const AS1 char* bPtr = (const AS1 char*)wsz +
                         ((size_t)ng * 4 * 1024 + (size_t)lane * 32);

  v8f c[4][4];                          // [m-subtile][n-subtile]
#pragma unroll
  for (int s = 0; s < 4; ++s)
#pragma unroll
    for (int nt = 0; nt < 4; ++nt)
#pragma unroll
      for (int r = 0; r < 8; ++r) c[s][nt][r] = 0.0f;

  for (int j = 0; j < BLOCK_SIZE; ++j) {        // row within block
#pragma unroll
    for (int qk = 0; qk < 4; ++qk) {            // K-chunk kc = j*4 + qk
      // ---- B fragments: 4 n-subtiles
      BU B[4];
      if (SWZ) {
#pragma unroll
        for (int nt = 0; nt < 4; ++nt) {
          const AS1 char* fp = bPtr + qk * 8192 + nt * 1024;
          B[nt].q[0] = *(const AS1 u32x4*)(fp);
          B[nt].q[1] = *(const AS1 u32x4*)(fp + 16);
        }
      } else {
#pragma unroll
        for (int nt = 0; nt < 4; ++nt) {
          int e = ng * 64 + nt * 16 + lrow;
          int kc = j * 4 + qk;
#pragma unroll
          for (int vi = 0; vi < 8; ++vi) {
            int t0 = (hi << 4) + (vi << 1);
            const AS1 float* p0 = w + (size_t)(kc * 32 + t0) * HEAD_DIM + e;
            B[nt].v[2 * vi]     = (__bf16)p0[0];
            B[nt].v[2 * vi + 1] = (__bf16)p0[HEAD_DIM];
          }
        }
      }

      // ---- A ping-pong: build subtile s+1 while WMMAs of subtile s run.
      v16bf Af[2];
      Af[0] = make_A(aPtr[0] + qk * 32);
#pragma unroll
      for (int s = 0; s < 4; ++s) {
        if (s < 3) Af[(s + 1) & 1] = make_A(aPtr[s + 1] + qk * 32);
#pragma unroll
        for (int nt = 0; nt < 4; ++nt)
          c[s][nt] = __builtin_amdgcn_wmma_f32_16x16x32_bf16(
              false, Af[s & 1], false, B[nt].v, (short)0, c[s][nt],
              false, false);
      }
    }
    // advance one k/v row (512 floats) and 4 K-chunks of B (32 KB)
#pragma unroll
    for (int s = 0; s < 4; ++s) aPtr[s] += NUM_HEADS * HEAD_DIM;
    bPtr += 4 * 8192;
  }

  // ---- Store per 32-bit C/D layout: lanes 0-15: M=r, N=lane;
  //      lanes 16-31: M=8+r, N=lane-16.
#pragma unroll
  for (int s = 0; s < 4; ++s) {
#pragma unroll
    for (int nt = 0; nt < 4; ++nt) {
#pragma unroll
      for (int r = 0; r < 8; ++r) {
        int m = s * 16 + (hi ? (8 + r) : r);
        int o = oBase + m;
        if (o < total_out) {
          out[((size_t)o * NUM_HEADS + h) * HEAD_DIM +
              ng * 64 + nt * 16 + lrow] = c[s][nt][r];
        }
      }
    }
  }
}

// ---------------------------------------------------------------------------
extern "C" void kernel_launch(void* const* d_in, const int* in_sizes, int n_in,
                              void* d_out, int out_size, void* d_ws, size_t ws_size,
                              hipStream_t stream) {
  const float* k  = (const float*)d_in[0];
  const float* v  = (const float*)d_in[1];
  const float* wk = (const float*)d_in[2];
  const float* wv = (const float*)d_in[3];
  const int*   cu = (const int*)d_in[4];

  const int nseq      = in_sizes[4] - 1;
  const int total_out = (out_size - (nseq + 1)) / (2 * NUM_HEADS * HEAD_DIM);

  float* out   = (float*)d_out;
  int* outTail = (int*)d_out + (size_t)2 * total_out * NUM_HEADS * HEAD_DIM;

  // scratch layout: base[total_out] ints, then swizzled bf16 weights (2 MB)
  char* ws = (char*)d_ws;
  size_t off = 0;
  int* base = (int*)(ws + off);
  off = (off + (size_t)total_out * sizeof(int) + 255) & ~(size_t)255;
  __bf16* wsW = (__bf16*)(ws + off);
  size_t need = off + (size_t)2 * SWZ_ELEMS * sizeof(__bf16);
  int useSwz = (ws_size >= need) ? 1 : 0;

  prep_kernel<<<1, 256, 0, stream>>>(cu, nseq, total_out, base, outTail);

  if (useSwz) {
    int tot = 2 * SWZ_ELEMS;
    swizzle_w_kernel<<<(tot + 255) / 256, 256, 0, stream>>>(wk, wv, wsW);
  }

  int mTiles = (total_out + 63) / 64;
  if (mTiles > 0) {
    dim3 grid(mTiles, NUM_HEADS, 4);   // z = which*2 + n-half
    if (useSwz)
      kv_compress_wmma<true><<<grid, 32, 0, stream>>>(k, v, wk, wv, base, wsW,
                                                      out, total_out);
    else
      kv_compress_wmma<false><<<grid, 32, 0, stream>>>(k, v, wk, wv, base, wsW,
                                                       out, total_out);
  }
}
